// StackedDecoder_54236847014270
// MI455X (gfx1250) — compile-verified, matrix operand order checked by reading, not run on hardware
//
#include <hip/hip_runtime.h>

#define Tt   12
#define Nn   10000
#define Ee   160000
#define Dd   128
#define OUTd 64
#define Ll   2

typedef __attribute__((ext_vector_type(16))) __bf16 v16bf;
typedef __attribute__((ext_vector_type(8)))  float  v8f;
typedef int v4i __attribute__((vector_size(16)));

#define AS1V(p) ((__attribute__((address_space(1))) v4i*)(p))
#define AS3V(p) ((__attribute__((address_space(3))) v4i*)(p))

__device__ __forceinline__ unsigned short f2bf(float f) {
    unsigned u = __builtin_bit_cast(unsigned, f);
    u += 0x7FFFu + ((u >> 16) & 1u);          // round-to-nearest-even
    return (unsigned short)(u >> 16);
}

// Async global->LDS copy of one 16-byte element per call; falls back to a
// synchronous VGPR round-trip copy when the gfx1250 async builtins are absent.
__device__ __forceinline__ void async_copy16(const void* g, void* l) {
#if __has_builtin(__builtin_amdgcn_global_load_async_to_lds_b128)
    __builtin_amdgcn_global_load_async_to_lds_b128(AS1V(g), AS3V(l), 0, 0);
#else
    *(uint4*)l = *(const uint4*)g;
#endif
}

__device__ __forceinline__ void async_wait0() {
#if __has_builtin(__builtin_amdgcn_global_load_async_to_lds_b128)
#if __has_builtin(__builtin_amdgcn_s_wait_asynccnt)
    __builtin_amdgcn_s_wait_asynccnt(0);
#else
    asm volatile("s_wait_asynccnt 0" ::: "memory");
#endif
#endif
}

// ---------------------------------------------------------------- utilities
__global__ void fill_zero(float* p, int n4) {
    int i = blockIdx.x * blockDim.x + threadIdx.x;
    if (i < n4) ((float4*)p)[i] = make_float4(0.f, 0.f, 0.f, 0.f);
}

__global__ void copy_f32(float* d, const float* s, int n4) {
    int i = blockIdx.x * blockDim.x + threadIdx.x;
    if (i < n4) ((float4*)d)[i] = ((const float4*)s)[i];
}

__global__ void deg_count(const int* __restrict__ dst, float* __restrict__ deg) {
    int e = blockIdx.x * blockDim.x + threadIdx.x;
    if (e < Ee) atomicAdd(deg + dst[e], 1.0f);
}

__global__ void deg_recip(const float* __restrict__ deg, float* __restrict__ dinv) {
    int i = blockIdx.x * blockDim.x + threadIdx.x;
    if (i < Nn) { float d = deg[i]; dinv[i] = d > 0.f ? 1.f / d : 0.f; }
}

// mean-aggregation numerator: acc[dst] += f[src]  (deg_inv fused into GEMM A-load)
__global__ void scatter_add(const float* __restrict__ f, const int* __restrict__ src,
                            const int* __restrict__ dst, float* __restrict__ acc) {
    int gid = blockIdx.x * blockDim.x + threadIdx.x;
    int e = gid >> 5;                      // 32 segments of 4 floats per edge
    if (e >= Ee) return;
    int dseg = (gid & 31) << 2;
    int s = src[e], d = dst[e];
    float4 v = *(const float4*)(f + (size_t)s * Dd + dseg);
    float* a = acc + (size_t)d * Dd + dseg;
    atomicAdd(a + 0, v.x); atomicAdd(a + 1, v.y);
    atomicAdd(a + 2, v.z); atomicAdd(a + 3, v.w);
}

// ------------------------------------------------------------ weight packing
// Wcat: [L][16 kchunk][16 ctile][32 lane][16 halfword] bf16, encodes the
// 512x256 concatenated weight (rows: xin|hN_x|h|hN_h ; cols: u-gate|c-gate)
// already in the 16-bit B-matrix (32x16) WMMA fragment layout.
__global__ void pack_wcat(const float* __restrict__ Wself, const float* __restrict__ Wneigh,
                          unsigned short* __restrict__ Wcat) {
    int gid = blockIdx.x * blockDim.x + threadIdx.x;
    if (gid >= Ll * 512 * 256) return;
    int l   = gid / (512 * 256);
    int rem = gid % (512 * 256);
    int k = rem >> 8;                      // 0..511
    int n = rem & 255;                     // 0..255
    int blk  = k >> 7;                     // 0:xin 1:hN_x 2:h 3:hN_h
    int kk   = k & 127;
    int gate = ((n < 128) ? 2 : 4) + (blk >> 1);   // u: nets 2,3 ; c: nets 4,5
    int d    = n & 127;
    const float* Wsrc = (blk & 1) ? Wneigh : Wself;
    float val = Wsrc[(((size_t)l * 6 + gate) * 128 + kk) * 128 + d];
    int kc = k >> 5, kin = k & 31;
    int ct = n >> 4;
    int hi = ((kin & 15) >= 8);
    int lanei = (n & 15) + (hi << 4);
    int idx = ((kin >> 4) << 3) + (kin & 7);
    Wcat[((((size_t)l * 16 + kc) * 16 + ct) * 32 + lanei) * 16 + idx] = f2bf(val);
}

// Wout: [4 kchunk][4 ctile][32 lane][16] bf16, same fragment layout.
__global__ void pack_wout(const float* __restrict__ Wo, unsigned short* __restrict__ Wop) {
    int gid = blockIdx.x * blockDim.x + threadIdx.x;
    if (gid >= Dd * OUTd) return;
    int k = gid / OUTd, col = gid % OUTd;
    int kc = k >> 5, kin = k & 31;
    int ct = col >> 4;
    int hi = ((kin & 15) >= 8);
    int lanei = (col & 15) + (hi << 4);
    int idx = ((kin >> 4) << 3) + (kin & 7);
    Wop[(((size_t)kc * 4 + ct) * 32 + lanei) * 16 + idx] = f2bf(Wo[(size_t)k * OUTd + col]);
}

__global__ void pack_bias(const float* __restrict__ b, float* __restrict__ bu,
                          float* __restrict__ bc) {
    int t = blockIdx.x * blockDim.x + threadIdx.x;
    if (t >= Ll * 2 * Dd) return;
    int l = t >> 8, kind = (t >> 7) & 1, d = t & 127;
    float v = b[(((size_t)l * 6) + (kind ? 4 : 2)) * 128 + d] +
              b[(((size_t)l * 6) + (kind ? 5 : 3)) * 128 + d];
    (kind ? bc : bu)[l * 128 + d] = v;
}

// ------------------------------------------------- fused layer GEMM + gating
// [64 rows x 512 K] @ [512 x 256] with K = concat(xin, hN_x, h, hN_h),
// cols 0..127 = u pre-act, 128..255 = c pre-act; then GRU-style update.
__global__ __launch_bounds__(256) void layer_gemm_gate(
    const float* __restrict__ xin, const float* __restrict__ hNx,
    const float* __restrict__ h,   const float* __restrict__ hNh,
    const float* __restrict__ dinv,
    const unsigned short* __restrict__ Wl,   // this layer's packed [16][16][32][16]
    const float* __restrict__ bu, const float* __restrict__ bc,
    float* __restrict__ hout) {
    __shared__ __attribute__((aligned(32))) unsigned short As[4][32][16];   //  4 KB
    __shared__ __attribute__((aligned(32))) unsigned short Bs[16][32][16];  // 16 KB
    __shared__ float preC[64][128];                                         // 32 KB

    const int tid = threadIdx.x;
    const int lane = tid & 31;
    const int wave = tid >> 5;
    const int rowstrip = wave & 3;     // 4 row strips of 16
    const int colhalf  = wave >> 2;    // 0: u-gate cols, 1: c-gate cols
    const int blockRow = blockIdx.x * 64;

    v8f acc[8] = {};

    const int arow = tid >> 2;         // staging: row 0..63
    const int aseg = tid & 3;          // 8-float segment within 32-wide chunk
    int agrow = blockRow + arow; if (agrow > Nn - 1) agrow = Nn - 1;
    const float nscale = dinv[agrow];
    const int rs = arow >> 4, rlo = arow & 15;
    const int k0 = aseg * 8;

    // per-thread swizzled LDS slots for the 4 packed pairs of the A stage
    unsigned* asl[4];
#pragma unroll
    for (int q = 0; q < 4; ++q) {
        int kk = k0 + q * 2;
        int hi = ((kk & 15) >= 8);
        int idx = ((kk >> 4) << 3) + (kk & 7);
        asl[q] = (unsigned*)&As[rs][rlo + (hi << 4)][idx];
    }

    const float* srcs[4] = { xin, hNx, h, hNh };

#pragma unroll
    for (int phase = 0; phase < 4; ++phase) {           // source fixed per phase
        const float* sp = srcs[phase];
        const float scale = (phase & 1) ? nscale : 1.0f;
        const float* pbase = sp + (size_t)agrow * Dd + k0;
#pragma unroll
        for (int kq = 0; kq < 4; ++kq) {
            const int kc = phase * 4 + kq;
            { // ---- stage A: fp32 load + deg_inv fuse + bf16 + fragment swizzle
                const float* p = pbase + kq * 32;
                float4 v0 = *(const float4*)p;
                float4 v1 = *(const float4*)(p + 4);
                float vv[8] = { v0.x, v0.y, v0.z, v0.w, v1.x, v1.y, v1.z, v1.w };
#pragma unroll
                for (int q = 0; q < 4; ++q) {
                    unsigned lo = f2bf(vv[q * 2] * scale);
                    unsigned hw = f2bf(vv[q * 2 + 1] * scale);
                    *asl[q] = lo | (hw << 16);
                }
            }
            { // ---- stage B: async copy of pre-swizzled bf16 weight chunk (16 KB)
                const uint4* gsrc = (const uint4*)(Wl + (size_t)kc * 8192);
                uint4* ldst = (uint4*)&Bs[0][0][0];
#pragma unroll
                for (int i = 0; i < 4; ++i)
                    async_copy16(gsrc + tid + i * 256, ldst + tid + i * 256);
            }
            if (kc + 1 < 16)  // warm caches for next weight chunk
                __builtin_prefetch(Wl + (size_t)(kc + 1) * 8192, 0, 0);
            async_wait0();
            __syncthreads();
            // ---- WMMA: one A frag per wave, 8 column tiles
            v16bf a = *(const v16bf*)&As[rowstrip][lane][0];
#pragma unroll
            for (int cc = 0; cc < 8; ++cc) {
                v16bf bm = *(const v16bf*)&Bs[colhalf * 8 + cc][lane][0];
                acc[cc] = __builtin_amdgcn_wmma_f32_16x16x32_bf16(
                    false, a, false, bm, (short)0, acc[cc], false, false);
            }
            __syncthreads();
        }
    }

    // ---- epilogue: c-half waves publish pre-acts; u-half waves gate + update h
    const int mlo = (lane >> 4) << 3;   // +0 / +8 row within tile
    const int nlo = lane & 15;
    if (colhalf == 1) {
#pragma unroll
        for (int cc = 0; cc < 8; ++cc)
#pragma unroll
            for (int r = 0; r < 8; ++r)
                preC[rowstrip * 16 + r + mlo][cc * 16 + nlo] = acc[cc][r];
    }
    __syncthreads();
    if (colhalf == 0) {
#pragma unroll
        for (int cc = 0; cc < 8; ++cc) {
#pragma unroll
            for (int r = 0; r < 8; ++r) {
                int row = rowstrip * 16 + r + mlo;
                int col = cc * 16 + nlo;
                int grow = blockRow + row;
                if (grow < Nn) {
                    float pu = acc[cc][r] + bu[col];
                    float pc = preC[row][col] + bc[col];
                    float u  = 1.f / (1.f + __expf(-pu));
                    float cg = tanhf(pc);
                    float hv = h[(size_t)grow * Dd + col];
                    hout[(size_t)grow * Dd + col] = u * hv + (1.f - u) * cg;
                }
            }
        }
    }
}

// ------------------------------------------------------- output projection
__global__ __launch_bounds__(256) void out_proj(
    const float* __restrict__ cur, const unsigned short* __restrict__ Wop,
    const float* __restrict__ bo, float* __restrict__ outp) {
    __shared__ __attribute__((aligned(32))) unsigned short As[4][32][16];
    __shared__ __attribute__((aligned(32))) unsigned short Bs[4][32][16];
    const int tid = threadIdx.x, lane = tid & 31, wave = tid >> 5;
    const int rowstrip = wave & 3, colpair = wave >> 2;
    const int blockRow = blockIdx.x * 64;
    v8f acc[2] = {};
    const int arow = tid >> 2, aseg = tid & 3;
    int agrow = blockRow + arow; if (agrow > Nn - 1) agrow = Nn - 1;
    const int rs = arow >> 4, rlo = arow & 15;
    const int k0 = aseg * 8;
#pragma unroll
    for (int kc = 0; kc < 4; ++kc) {
        {
            const float* p = cur + (size_t)agrow * Dd + kc * 32 + k0;
            float4 v0 = *(const float4*)p;
            float4 v1 = *(const float4*)(p + 4);
            float vv[8] = { v0.x, v0.y, v0.z, v0.w, v1.x, v1.y, v1.z, v1.w };
#pragma unroll
            for (int q = 0; q < 4; ++q) {
                int kk = k0 + q * 2;
                int hi = ((kk & 15) >= 8);
                int idx = ((kk >> 4) << 3) + (kk & 7);
                unsigned lo = f2bf(vv[q * 2]);
                unsigned hw = f2bf(vv[q * 2 + 1]);
                *(unsigned*)&As[rs][rlo + (hi << 4)][idx] = lo | (hw << 16);
            }
        }
        async_copy16((const uint4*)(Wop + (size_t)kc * 2048) + tid,
                     (uint4*)&Bs[0][0][0] + tid);
        async_wait0();
        __syncthreads();
        v16bf a = *(const v16bf*)&As[rowstrip][lane][0];
#pragma unroll
        for (int j = 0; j < 2; ++j) {
            v16bf bm = *(const v16bf*)&Bs[colpair * 2 + j][lane][0];
            acc[j] = __builtin_amdgcn_wmma_f32_16x16x32_bf16(
                false, a, false, bm, (short)0, acc[j], false, false);
        }
        __syncthreads();
    }
    const int mlo = (lane >> 4) << 3, nlo = lane & 15;
#pragma unroll
    for (int j = 0; j < 2; ++j)
#pragma unroll
        for (int r = 0; r < 8; ++r) {
            int grow = blockRow + rowstrip * 16 + r + mlo;
            int col = (colpair * 2 + j) * 16 + nlo;
            if (grow < Nn)
                outp[(size_t)grow * OUTd + col] = acc[j][r] + bo[col];
        }
}

// --------------------------------------------------------------------------
extern "C" void kernel_launch(void* const* d_in, const int* in_sizes, int n_in,
                              void* d_out, int out_size, void* d_ws, size_t ws_size,
                              hipStream_t stream) {
    (void)in_sizes; (void)n_in; (void)out_size; (void)ws_size;
    const float* x      = (const float*)d_in[0];  // [T,N,D]
    const float* hin    = (const float*)d_in[1];  // [L,N,D]
    const float* Wself  = (const float*)d_in[2];  // [L,6,D,D]
    const float* Wneigh = (const float*)d_in[3];  // [L,6,D,D]
    const float* b      = (const float*)d_in[4];  // [L,6,D]
    const float* Wo     = (const float*)d_in[5];  // [D,OUT]
    const float* bo     = (const float*)d_in[6];  // [OUT]
    const int*   src    = (const int*)d_in[7];    // [E]
    const int*   dst    = (const int*)d_in[8];    // [E]
    float* out = (float*)d_out;                   // [T,N,OUT] ++ [L,N,D]

    char* w = (char*)d_ws;
    float* hid  = (float*)w; w += (size_t)Ll * Nn * Dd * 4;
    float* accA = (float*)w; w += (size_t)Nn * Dd * 4;
    float* accB = (float*)w; w += (size_t)Nn * Dd * 4;
    float* deg  = (float*)w; w += (size_t)Nn * 4;
    float* dinv = (float*)w; w += (size_t)Nn * 4;
    float* bu   = (float*)w; w += (size_t)Ll * Dd * 4;
    float* bc   = (float*)w; w += (size_t)Ll * Dd * 4;
    unsigned short* Wcat = (unsigned short*)w; w += (size_t)Ll * 512 * 256 * 2;
    unsigned short* Wop  = (unsigned short*)w; w += (size_t)Dd * OUTd * 2;

    const int B = 256;
    // one-time (per launch, deterministic) setup
    fill_zero<<<(Nn / 4 + B - 1) / B, B, 0, stream>>>(deg, Nn / 4);
    deg_count<<<(Ee + B - 1) / B, B, 0, stream>>>(dst, deg);
    deg_recip<<<(Nn + B - 1) / B, B, 0, stream>>>(deg, dinv);
    pack_wcat<<<(Ll * 512 * 256) / B, B, 0, stream>>>(Wself, Wneigh, Wcat);
    pack_wout<<<(Dd * OUTd) / B, B, 0, stream>>>(Wo, Wop);
    pack_bias<<<(Ll * 2 * Dd + B - 1) / B, B, 0, stream>>>(b, bu, bc);
    copy_f32<<<((Ll * Nn * Dd / 4) + B - 1) / B, B, 0, stream>>>(hid, hin, Ll * Nn * Dd / 4);

    const int ND4   = Nn * Dd / 4;
    const int gz    = (ND4 + B - 1) / B;
    const int gsc   = (Ee * 32 + B - 1) / B;
    const int gtile = (Nn + 63) / 64;

    for (int t = 0; t < Tt; ++t) {
        const float* xt = x + (size_t)t * Nn * Dd;
        float* h0 = hid;
        float* h1 = hid + (size_t)Nn * Dd;
        // ---- layer 0: hN_x = agg(x_t), hN_h = agg(h0)
        fill_zero<<<gz, B, 0, stream>>>(accA, ND4);
        scatter_add<<<gsc, B, 0, stream>>>(xt, src, dst, accA);
        fill_zero<<<gz, B, 0, stream>>>(accB, ND4);
        scatter_add<<<gsc, B, 0, stream>>>(h0, src, dst, accB);
        layer_gemm_gate<<<gtile, B, 0, stream>>>(xt, accA, h0, accB, dinv,
                                                 Wcat, bu, bc, h0);
        // ---- layer 1: hN_x = agg(cur0), hN_h = agg(h1)
        fill_zero<<<gz, B, 0, stream>>>(accA, ND4);
        scatter_add<<<gsc, B, 0, stream>>>(h0, src, dst, accA);
        fill_zero<<<gz, B, 0, stream>>>(accB, ND4);
        scatter_add<<<gsc, B, 0, stream>>>(h1, src, dst, accB);
        layer_gemm_gate<<<gtile, B, 0, stream>>>(h0, accA, h1, accB, dinv,
                                                 Wcat + (size_t)16 * 8192,
                                                 bu + Dd, bc + Dd, h1);
        // ---- output projection
        out_proj<<<gtile, B, 0, stream>>>(h1, Wop, bo, out + (size_t)t * Nn * OUTd);
    }
    // hidden_final appended after outputs
    copy_f32<<<((Ll * Nn * Dd / 4) + B - 1) / B, B, 0, stream>>>(
        out + (size_t)Tt * Nn * OUTd, hid, Ll * Nn * Dd / 4);
}
